// ProbAttention_60945585930286
// MI455X (gfx1250) — matrix operand order repaired and checked by reading, not compiled, not run on hardware
//
#include <hip/hip_runtime.h>
#include <hip/hip_bf16.h>

#define BB 2
#define LL 2048
#define DD 512
#define HH 8
#define HDD 64
#define NROW (BB*LL)   // 4096

typedef __attribute__((ext_vector_type(16))) __bf16 v16bf;
typedef __attribute__((ext_vector_type(8)))  __bf16 v8bf;
typedef __attribute__((ext_vector_type(8)))  float  v8f;

#define WMMA_BF16(A,B,C) __builtin_amdgcn_wmma_f32_16x16x32_bf16(false,(A),false,(B),(short)0,(C),false,false)

// Build a 16-element bf16 A/B operand from two contiguous 8-element (16B) chunks.
static __device__ __forceinline__ v16bf ld16(const __bf16* p0, const __bf16* p1){
  v8bf lo = *(const v8bf*)p0;
  v8bf hi = *(const v8bf*)p1;
  v16bf r;
#pragma unroll
  for(int i=0;i<8;i++){ r[i]=lo[i]; r[i+8]=hi[i]; }
  return r;
}

// reductions across the 16-lane half-groups (wave32): every lane gets its row's value
static __device__ __forceinline__ float rowmax16(float v){
  v = fmaxf(v, __shfl_xor(v,1,32));
  v = fmaxf(v, __shfl_xor(v,2,32));
  v = fmaxf(v, __shfl_xor(v,4,32));
  v = fmaxf(v, __shfl_xor(v,8,32));
  return v;
}
static __device__ __forceinline__ float rowsum16(float v){
  v += __shfl_xor(v,1,32); v += __shfl_xor(v,2,32);
  v += __shfl_xor(v,4,32); v += __shfl_xor(v,8,32);
  return v;
}

// ---------------- fp32 -> bf16 convert (8 elements / thread) ----------------
__global__ __launch_bounds__(256) void cvt_kernel(const float* __restrict__ src,
                                                  __bf16* __restrict__ dst, int n){
  int base = (blockIdx.x*256 + threadIdx.x)*8;
  if(base < n){
    float4 f0 = *(const float4*)(src + base);
    float4 f1 = *(const float4*)(src + base + 4);
    v8bf o;
    o[0]=(__bf16)f0.x; o[1]=(__bf16)f0.y; o[2]=(__bf16)f0.z; o[3]=(__bf16)f0.w;
    o[4]=(__bf16)f1.x; o[5]=(__bf16)f1.y; o[6]=(__bf16)f1.z; o[7]=(__bf16)f1.w;
    *(v8bf*)(dst + base) = o;
  }
}

// ---------------- GEMM: Y = X * W^T + b  (32x64 tile / wave, double-buffered) --
// X: [NROW, DD] bf16 row-major, W: [DD, DD] bf16 row-major (output col = W row).
// mode 0: bf16 row-major [NROW, DD]
// mode 1: bf16 transposed  out[(b*DD + col)*LL + l]   (V^T layout [B,H,HD,L])
// mode 2: fp32 row-major (final output)
__global__ __launch_bounds__(128)
void gemm_kernel(const __bf16* __restrict__ X, const __bf16* __restrict__ W,
                 const float* __restrict__ bias, void* __restrict__ out, int mode){
  const int wave = blockIdx.x*4 + (threadIdx.x>>5);
  const int lane = threadIdx.x & 31;
  const int l15  = lane & 15;
  const int lhalf= lane >> 4;            // 0: lanes 0-15, 1: lanes 16-31
  const int mtile= wave >> 3;            // 128 row tiles of 32
  const int cg   = wave & 7;             // 8 col groups of 64
  const int rowbase = mtile*32;
  const int colbase = cg*64;

  v8f acc[2][4];
#pragma unroll
  for(int mt=0;mt<2;mt++)
#pragma unroll
    for(int nb=0;nb<4;nb++)
#pragma unroll
      for(int i=0;i<8;i++) acc[mt][nb][i] = 0.f;

  const __bf16* xr0 = X + (size_t)(rowbase +      l15)*DD;
  const __bf16* xr1 = X + (size_t)(rowbase + 16 + l15)*DD;
  const __bf16* wrb = W + (size_t)(colbase + l15)*DD + lhalf*16;

  // prologue: load operands for kk = 0
  v16bf a0 = ld16(xr0 + lhalf*16, xr0 + lhalf*16 + 8);
  v16bf a1 = ld16(xr1 + lhalf*16, xr1 + lhalf*16 + 8);
  v16bf b0 = ld16(wrb +  0*16*DD, wrb +  0*16*DD + 8);
  v16bf b1 = ld16(wrb +  1*16*DD, wrb +  1*16*DD + 8);
  v16bf b2 = ld16(wrb +  2*16*DD, wrb +  2*16*DD + 8);
  v16bf b3 = ld16(wrb +  3*16*DD, wrb +  3*16*DD + 8);

  for(int kk=0; kk<DD; kk+=32){
    // prefetch next step's operands into a second register set (redundant
    // reload of step 0 on the final iteration keeps the loop branch-free)
    const int kn = (kk+32 < DD) ? kk+32 : 0;
    v16bf na0 = ld16(xr0 + kn + lhalf*16, xr0 + kn + lhalf*16 + 8);
    v16bf na1 = ld16(xr1 + kn + lhalf*16, xr1 + kn + lhalf*16 + 8);
    v16bf nb0 = ld16(wrb + 0*16*DD + kn, wrb + 0*16*DD + kn + 8);
    v16bf nb1 = ld16(wrb + 1*16*DD + kn, wrb + 1*16*DD + kn + 8);
    v16bf nb2 = ld16(wrb + 2*16*DD + kn, wrb + 2*16*DD + kn + 8);
    v16bf nb3 = ld16(wrb + 3*16*DD + kn, wrb + 3*16*DD + kn + 8);

    acc[0][0] = WMMA_BF16(a0, b0, acc[0][0]);
    acc[1][0] = WMMA_BF16(a1, b0, acc[1][0]);
    acc[0][1] = WMMA_BF16(a0, b1, acc[0][1]);
    acc[1][1] = WMMA_BF16(a1, b1, acc[1][1]);
    acc[0][2] = WMMA_BF16(a0, b2, acc[0][2]);
    acc[1][2] = WMMA_BF16(a1, b2, acc[1][2]);
    acc[0][3] = WMMA_BF16(a0, b3, acc[0][3]);
    acc[1][3] = WMMA_BF16(a1, b3, acc[1][3]);

    a0 = na0; a1 = na1; b0 = nb0; b1 = nb1; b2 = nb2; b3 = nb3;
  }

#pragma unroll
  for(int mt=0;mt<2;mt++)
#pragma unroll
  for(int nb=0;nb<4;nb++){
    int col = colbase + nb*16 + l15;
    float bv = bias[col];
#pragma unroll
    for(int r=0;r<8;r++){
      int row = rowbase + mt*16 + r + lhalf*8;   // C layout: VGPR r -> row r / r+8
      float v = acc[mt][nb][r] + bv;
      if(mode == 0){
        ((__bf16*)out)[(size_t)row*DD + col] = (__bf16)v;
      } else if(mode == 1){
        int bb = row >> 11, l = row & (LL-1);
        ((__bf16*)out)[((size_t)bb*DD + col)*LL + l] = (__bf16)v;
      } else {
        ((float*)out)[(size_t)row*DD + col] = v;
      }
    }
  }
}

// ---------------- Vsum[b,h,d] = sum_k V[k][d]  (from V^T rows) ----------------
__global__ __launch_bounds__(256) void vsum_kernel(const __bf16* __restrict__ Vt,
                                                   float* __restrict__ Vsum){
  int i = blockIdx.x*256 + threadIdx.x;      // 0..1023 rows of Vt
  if(i < BB*HH*HDD){
    const __bf16* p = Vt + (size_t)i*LL;
    float s = 0.f;
    for(int l=0; l<LL; l+=8){
      v8bf v = *(const v8bf*)(p + l);
#pragma unroll
      for(int j=0;j<8;j++) s += (float)v[j];
    }
    Vsum[i] = s;
  }
}

// ---------------- fused double-softmax attention ----------------
// Qh,Kh: bf16 [NROW, DD] (== [B,L,H,HD]); Vt: bf16 [B,H,HD,L]; pm: bool bytes [B,H,L,L]
// X2: bf16 [NROW, DD] attention output (pre output-projection)
__global__ __launch_bounds__(128)
void attn_kernel(const __bf16* __restrict__ Qh, const __bf16* __restrict__ Kh,
                 const __bf16* __restrict__ Vt, const float* __restrict__ Vsum,
                 const unsigned char* __restrict__ pm, __bf16* __restrict__ X2){
  __shared__ __align__(16) __bf16 Plds[4][2][16*32];

  const int wid  = threadIdx.x >> 5;
  const int lane = threadIdx.x & 31;
  const int l15  = lane & 15;
  const int lhalf= lane >> 4;
  const int gw   = blockIdx.x*4 + wid;       // 2048 q-tiles total
  const int qt   = gw & 127;
  const int bh   = gw >> 7;                  // 0..15
  const int h    = bh & 7;
  const int b    = bh >> 3;
  const int qb   = qt * 16;
  const float scale = 0.125f;                // 1/sqrt(64)

  // Q tile (16 rows x 64) in A layout, for both K-chunks of the dot over HD=64
  const __bf16* qrow = Qh + (size_t)(b*LL + qb + l15)*DD + h*HDD;
  v16bf aq0 = ld16(qrow +  0 + lhalf*16, qrow +  8 + lhalf*16);
  v16bf aq1 = ld16(qrow + 32 + lhalf*16, qrow + 40 + lhalf*16);

  const __bf16* kbase = Kh + (size_t)(b*LL + l15)*DD + h*HDD;
  const int kend = qb + 16;                  // causal cols are [0, qb+15]

  // ---- pass 1: online softmax stats (m1, s1); K tiles double-buffered
  float m1[8], s1[8];
#pragma unroll
  for(int r=0;r<8;r++){ m1[r] = -__builtin_inff(); s1[r] = 0.f; }

  v16bf bk0 = ld16(kbase +      lhalf*16, kbase +      lhalf*16 + 8);
  v16bf bk1 = ld16(kbase + 32 + lhalf*16, kbase + 32 + lhalf*16 + 8);
  for(int kb=0; kb<kend; kb+=16){
    v16bf c0 = bk0, c1 = bk1;
    const int knext = (kb+16 < kend) ? kb+16 : 0;   // harmless redundant reload
    const __bf16* kr = kbase + (size_t)knext*DD;
    bk0 = ld16(kr +      lhalf*16, kr +      lhalf*16 + 8);
    bk1 = ld16(kr + 32 + lhalf*16, kr + 32 + lhalf*16 + 8);

    v8f c;
#pragma unroll
    for(int i=0;i<8;i++) c[i] = 0.f;
    c = WMMA_BF16(aq0, c0, c);
    c = WMMA_BF16(aq1, c1, c);
#pragma unroll
    for(int r=0;r<8;r++){
      int row = qb + r + lhalf*8;
      int col = kb + l15;
      float v = (col <= row) ? c[r]*scale : -__builtin_inff();
      float nm = fmaxf(m1[r], rowmax16(v));
      float ts = rowsum16(__expf(v - nm));
      s1[r] = s1[r]*__expf(m1[r] - nm) + ts;
      m1[r] = nm;
    }
  }
  float inv1[8];
#pragma unroll
  for(int r=0;r<8;r++) inv1[r] = 1.f / s1[r];

  // ---- pass 2: second softmax over z = pm*aw, flash-style with accumulators
  float m2[8], s2[8], n1[8];
#pragma unroll
  for(int r=0;r<8;r++){ m2[r] = 0.f; s2[r] = 0.f; n1[r] = 0.f; }
  v8f acc1[4], accV[4];
#pragma unroll
  for(int nb=0;nb<4;nb++)
#pragma unroll
    for(int i=0;i<8;i++){ acc1[nb][i] = 0.f; accV[nb][i] = 0.f; }

  const __bf16* vbase = Vt + (size_t)bh*HDD*LL;
  __bf16* Pw  = &Plds[wid][0][0];
  __bf16* P0w = &Plds[wid][1][0];

  for(int kb=0; kb<kend; kb+=32){
    // hoist V-tile loads: they complete while we compute the stats below
    v16bf bv[4];
#pragma unroll
    for(int nb=0;nb<4;nb++){
      const __bf16* vr = vbase + (size_t)(nb*16 + l15)*LL + kb + lhalf*16;
      bv[nb] = ld16(vr, vr+8);
    }

    float zarr[2][8], pmarr[2][8];
#pragma unroll
    for(int t=0;t<2;t++){
      int kt = kb + t*16;
      if(kt < kend){                                  // wave-uniform branch
        const __bf16* kr = kbase + (size_t)kt*DD;
        v16bf c0 = ld16(kr +      lhalf*16, kr +      lhalf*16 + 8);
        v16bf c1 = ld16(kr + 32 + lhalf*16, kr + 32 + lhalf*16 + 8);
        v8f c;
#pragma unroll
        for(int i=0;i<8;i++) c[i] = 0.f;
        c = WMMA_BF16(aq0, c0, c);
        c = WMMA_BF16(aq1, c1, c);
#pragma unroll
        for(int r=0;r<8;r++){
          int row = qb + r + lhalf*8;
          int col = kt + l15;
          float v = (col <= row) ? c[r]*scale : -__builtin_inff();
          float aw = __expf(v - m1[r]) * inv1[r];     // first-softmax prob
          float p  = (float)pm[((size_t)bh*LL + row)*LL + col];
          pmarr[t][r] = p;
          zarr[t][r]  = p * aw;
        }
      } else {
#pragma unroll
        for(int r=0;r<8;r++){ pmarr[t][r] = 0.f; zarr[t][r] = 0.f; }
      }
    }
    float frow[8];
#pragma unroll
    for(int r=0;r<8;r++){
      float zm = rowmax16(fmaxf(zarr[0][r], zarr[1][r]));
      float nm = fmaxf(m2[r], zm);
      float f  = __expf(m2[r] - nm);
      float te0 = pmarr[0][r]*__expf(zarr[0][r] - nm);
      float te1 = pmarr[1][r]*__expf(zarr[1][r] - nm);
      s2[r] = s2[r]*f + rowsum16(te0 + te1);
      n1[r] += rowsum16(pmarr[0][r] + pmarr[1][r]);
      m2[r] = nm; frow[r] = f;
      int lrow = r + lhalf*8;
      Pw [lrow*32 +      l15] = (__bf16)te0;          // C-layout -> LDS row-major
      Pw [lrow*32 + 16 + l15] = (__bf16)te1;
      P0w[lrow*32 +      l15] = (__bf16)pmarr[0][r];
      P0w[lrow*32 + 16 + l15] = (__bf16)pmarr[1][r];
    }
#pragma unroll
    for(int nb=0;nb<4;nb++)
#pragma unroll
      for(int i=0;i<8;i++) acc1[nb][i] *= frow[i];

    asm volatile("s_wait_dscnt 0" ::: "memory");      // cross-lane LDS visibility

    // reload P tiles in A layout (16x32)
    const __bf16* pp  = Pw  + l15*32;
    const __bf16* pp0 = P0w + l15*32;
    v16bf aP  = ld16(pp  + lhalf*16, pp  + lhalf*16 + 8);
    v16bf aP0 = ld16(pp0 + lhalf*16, pp0 + lhalf*16 + 8);
#pragma unroll
    for(int nb=0;nb<4;nb++){
      acc1[nb] = WMMA_BF16(aP,  bv[nb], acc1[nb]);    // sum_{pm=1} e^{aw-m2} V
      accV[nb] = WMMA_BF16(aP0, bv[nb], accV[nb]);    // sum_{pm=1} V
    }
  }

  // ---- finalize: out = (acc1 + e^{-m2}(Vsum - accV)) / (s2 + (L-n1)e^{-m2})
  float e2r[8], invd[8];
#pragma unroll
  for(int r=0;r<8;r++){
    e2r[r]  = __expf(-m2[r]);
    invd[r] = 1.f / (s2[r] + ((float)LL - n1[r])*e2r[r]);
  }
#pragma unroll
  for(int nb=0;nb<4;nb++){
    int d = nb*16 + l15;
    float vs = Vsum[bh*HDD + d];
#pragma unroll
    for(int r=0;r<8;r++){
      int row = qb + r + lhalf*8;
      float o = (acc1[nb][r] + e2r[r]*(vs - accV[nb][r])) * invd[r];
      X2[(size_t)(b*LL + row)*DD + h*HDD + d] = (__bf16)o;
    }
  }
}

extern "C" void kernel_launch(void* const* d_in, const int* in_sizes, int n_in,
                              void* d_out, int out_size, void* d_ws, size_t ws_size,
                              hipStream_t stream){
  (void)in_sizes; (void)n_in; (void)out_size; (void)ws_size;
  const float* q  = (const float*)d_in[0];
  const float* Wq = (const float*)d_in[1];
  const float* bq = (const float*)d_in[2];
  const float* Wk = (const float*)d_in[3];
  const float* bk = (const float*)d_in[4];
  const float* Wv = (const float*)d_in[5];
  const float* bv = (const float*)d_in[6];
  const float* Wo = (const float*)d_in[7];
  const float* bo = (const float*)d_in[8];
  const unsigned char* pm = (const unsigned char*)d_in[9];  // jax bool = 1 byte

  char* ws = (char*)d_ws;
  size_t off = 0;
  auto walloc = [&](size_t bytes)->void*{
    void* p = ws + off; off += (bytes + 255) & ~(size_t)255; return p;
  };
  __bf16* qbf = (__bf16*)walloc((size_t)NROW*DD*2);
  __bf16* wqb = (__bf16*)walloc((size_t)DD*DD*2);
  __bf16* wkb = (__bf16*)walloc((size_t)DD*DD*2);
  __bf16* wvb = (__bf16*)walloc((size_t)DD*DD*2);
  __bf16* wob = (__bf16*)walloc((size_t)DD*DD*2);
  __bf16* Qh  = (__bf16*)walloc((size_t)NROW*DD*2);
  __bf16* Kh  = (__bf16*)walloc((size_t)NROW*DD*2);
  __bf16* Vt  = (__bf16*)walloc((size_t)BB*HH*HDD*LL*2);
  __bf16* X2  = (__bf16*)walloc((size_t)NROW*DD*2);
  float*  Vs  = (float*) walloc((size_t)BB*HH*HDD*4);

  const int nq = NROW*DD, nw = DD*DD;
  cvt_kernel<<<(nq/8+255)/256, 256, 0, stream>>>(q,  qbf, nq);
  cvt_kernel<<<(nw/8+255)/256, 256, 0, stream>>>(Wq, wqb, nw);
  cvt_kernel<<<(nw/8+255)/256, 256, 0, stream>>>(Wk, wkb, nw);
  cvt_kernel<<<(nw/8+255)/256, 256, 0, stream>>>(Wv, wvb, nw);
  cvt_kernel<<<(nw/8+255)/256, 256, 0, stream>>>(Wo, wob, nw);

  // 1024 waves per GEMM (32x64 tiles), 4 waves/block
  gemm_kernel<<<256, 128, 0, stream>>>(qbf, wqb, bq, Qh, 0);
  gemm_kernel<<<256, 128, 0, stream>>>(qbf, wkb, bk, Kh, 0);
  gemm_kernel<<<256, 128, 0, stream>>>(qbf, wvb, bv, Vt, 1);

  vsum_kernel<<<4, 256, 0, stream>>>(Vt, Vs);

  attn_kernel<<<512, 128, 0, stream>>>(Qh, Kh, Vt, Vs, pm, X2);

  gemm_kernel<<<256, 128, 0, stream>>>(X2, wob, bo, d_out, 2);
}
